// MultiHeadAttentionSplit_27393301414302
// MI455X (gfx1250) — matrix operand order repaired and checked by reading, not compiled, not run on hardware
//
#include <hip/hip_runtime.h>
#include <stdint.h>

// Problem constants (match reference)
#define S_LEN  2048
#define BATCH  2
#define DMODEL 1024
#define HEADS  16
#define DKH    64
#define ROWS   (S_LEN * BATCH)        // 4096 GEMM rows

typedef __attribute__((ext_vector_type(16))) __bf16   bf16x16;
typedef __attribute__((ext_vector_type(8)))  float    f32x8;
typedef __attribute__((ext_vector_type(8)))  uint32_t u32x8;

__device__ __forceinline__ uint16_t f32_to_bf16(float f) {
    uint32_t u = __builtin_bit_cast(uint32_t, f);
    uint32_t r = u + 0x7FFFu + ((u >> 16) & 1u);   // round-to-nearest-even
    return (uint16_t)(r >> 16);
}
__device__ __forceinline__ float bf16_to_f32(uint16_t u) {
    return __builtin_bit_cast(float, (uint32_t)u << 16);
}

// Build a v16bf WMMA fragment from two 16-byte chunks (generic ptr: global or LDS)
__device__ __forceinline__ bf16x16 make_frag(const uint16_t* p0, const uint16_t* p1) {
    uint4 lo = *(const uint4*)p0;
    uint4 hi = *(const uint4*)p1;
    u32x8 v = {lo.x, lo.y, lo.z, lo.w, hi.x, hi.y, hi.z, hi.w};
    return __builtin_bit_cast(bf16x16, v);
}

__device__ __forceinline__ f32x8 wmma_bf16(bf16x16 a, bf16x16 b, f32x8 c) {
    return __builtin_amdgcn_wmma_f32_16x16x32_bf16(false, a, false, b, (short)0, c, false, false);
}

// ---------------------------------------------------------------- cast f32 -> bf16
__global__ void cast_f32_to_bf16(const float* __restrict__ src, uint16_t* __restrict__ dst, int n) {
    for (int i = blockIdx.x * blockDim.x + threadIdx.x; i < n; i += gridDim.x * blockDim.x)
        dst[i] = f32_to_bf16(src[i]);
}

// ---------------------------------------------------------------- zero f32
__global__ void zero_f32(float* __restrict__ dst, int n) {
    for (int i = blockIdx.x * blockDim.x + threadIdx.x; i < n; i += gridDim.x * blockDim.x)
        dst[i] = 0.0f;
}

// ------------------------------------------------- V transpose: [s,b,e] -> [b,h,dk,s]
__global__ void transpose_v(const uint16_t* __restrict__ Vb, uint16_t* __restrict__ Vt) {
    const int total = S_LEN * BATCH * DMODEL;
    for (int i = blockIdx.x * blockDim.x + threadIdx.x; i < total; i += gridDim.x * blockDim.x) {
        int e = i % DMODEL;
        int b = (i / DMODEL) % BATCH;
        int s = i / (DMODEL * BATCH);
        int h = e / DKH, dk = e % DKH;
        Vt[(((size_t)b * HEADS + h) * DKH + dk) * S_LEN + s] = Vb[i];
    }
}

// ------------------------------------------------- GEMM: C[r,n] = sum_k A[r,k] * W[n,k]
// A: M x K bf16 row-major, W: N x K bf16 row-major (torch Linear convention).
// Block = 128 threads (4 waves, 2x2); each wave computes a 64x64 tile:
// 16 WMMA accumulators (128 VGPRs), 4 A-frags + 4 B-frags per K-step -> 1:1 load:WMMA.
__global__ __launch_bounds__(128)
void gemm_bf16_wmma(const uint16_t* __restrict__ A, const uint16_t* __restrict__ W,
                    uint16_t* __restrict__ Cb, float* __restrict__ Cf,
                    int M, int N, int K, int out_f32)
{
    const int lane  = threadIdx.x & 31;
    const int wave  = threadIdx.x >> 5;
    const int wm    = wave >> 1;            // 0..1
    const int wn    = wave & 1;             // 0..1
    const int m_base = blockIdx.x * 128 + wm * 64;
    const int n_base = blockIdx.y * 128 + wn * 64;
    const int mrow  = lane & 15;
    const int khalf = lane >> 4;

    f32x8 acc[4][4] = {};

    for (int k0 = 0; k0 < K; k0 += 32) {
        bf16x16 afrag[4], bfrag[4];
#pragma unroll
        for (int mi = 0; mi < 4; ++mi) {
            const uint16_t* p = A + (size_t)(m_base + mi * 16 + mrow) * K + k0 + khalf * 8;
            afrag[mi] = make_frag(p, p + 16);           // K: {kb..kb+7, kb+16..kb+23}
        }
#pragma unroll
        for (int ni = 0; ni < 4; ++ni) {
            const uint16_t* p = W + (size_t)(n_base + ni * 16 + mrow) * K + k0 + khalf * 8;
            bfrag[ni] = make_frag(p, p + 16);           // K: {kb..kb+7, kb+16..kb+23}
        }
        if (k0 + 32 < K) {
            __builtin_prefetch(A + (size_t)(m_base + mrow) * K + k0 + 32, 0, 1);
            __builtin_prefetch(W + (size_t)(n_base + mrow) * K + k0 + 32, 0, 1);
        }
#pragma unroll
        for (int mi = 0; mi < 4; ++mi)
#pragma unroll
            for (int ni = 0; ni < 4; ++ni)
                acc[mi][ni] = wmma_bf16(afrag[mi], bfrag[ni], acc[mi][ni]);
    }

#pragma unroll
    for (int mi = 0; mi < 4; ++mi)
#pragma unroll
        for (int ni = 0; ni < 4; ++ni)
#pragma unroll
            for (int v = 0; v < 8; ++v) {
                int row = m_base + mi * 16 + khalf * 8 + v;
                int col = n_base + ni * 16 + mrow;
                float val = acc[mi][ni][v];
                if (out_f32) Cf[(size_t)row * N + col] = val;
                else         Cb[(size_t)row * N + col] = f32_to_bf16(val);
            }
}

// ------------------------------------------------- fused attention
// 4 waves per block, each wave owns one q-tile of 16 rows for one (b,h).
// LDS: per-wave bf16 strip [16][S_LEN] = 64KB -> 256KB/block (< 320KB/WGP),
// so all 4 SIMD32s of the WGP stay busy. Scores stored bf16, softmax stats in
// f32 registers, normalized probs written back in place (each lane touches
// only its own half-row -> no cross-lane hazard), then fed to P@V WMMA via
// ds_load_b128 A-fragments.
__global__ __launch_bounds__(128)
void attention_wmma(const uint16_t* __restrict__ Qb, const uint16_t* __restrict__ Kb,
                    const uint16_t* __restrict__ Vt, uint16_t* __restrict__ Ctx,
                    float* __restrict__ AttnMean)
{
    extern __shared__ unsigned char smem_raw[];
    const int lane  = threadIdx.x & 31;
    const int wave  = threadIdx.x >> 5;
    const int mrow  = lane & 15;
    const int khalf = lane >> 4;

    uint16_t* sc = (uint16_t*)smem_raw + (size_t)wave * 16 * S_LEN;   // [16][S_LEN] bf16

    const int bh = blockIdx.y;
    const int b  = bh / HEADS;
    const int h  = bh % HEADS;
    const int q0 = (blockIdx.x * 4 + wave) * 16;

    // Q A-fragments (dk 0..31 and 32..63), loaded once
    bf16x16 qf[2];
#pragma unroll
    for (int kd = 0; kd < 2; ++kd) {
        const uint16_t* p = Qb + ((size_t)(q0 + mrow) * BATCH + b) * DMODEL
                               + h * DKH + kd * 32 + khalf * 8;
        qf[kd] = make_frag(p, p + 16);
    }

    const float scale = 0.125f;   // 1/sqrt(64)

    // ---- scores = (Q K^T) * scale, streamed into LDS as bf16
    for (int kc = 0; kc < S_LEN / 16; ++kc) {
        f32x8 c = {};
        const int scol = kc * 16 + mrow;
#pragma unroll
        for (int kd = 0; kd < 2; ++kd) {
            const uint16_t* p = Kb + ((size_t)scol * BATCH + b) * DMODEL
                                   + h * DKH + kd * 32 + khalf * 8;
            bf16x16 bf = make_frag(p, p + 16);
            c = wmma_bf16(qf[kd], bf, c);
        }
#pragma unroll
        for (int v = 0; v < 8; ++v)
            sc[(khalf * 8 + v) * S_LEN + kc * 16 + mrow] = f32_to_bf16(c[v] * scale);
    }
    __syncthreads();

    // ---- softmax per row; 2 lanes per row, halves combined via wave32 shfl_xor(16)
    {
        const int r  = lane & 15;
        const int c0 = (lane >> 4) * (S_LEN / 2);
        uint16_t* row = sc + (size_t)r * S_LEN + c0;

        float mx = -3.0e38f;
        for (int j = 0; j < S_LEN / 2; ++j) mx = fmaxf(mx, bf16_to_f32(row[j]));
        mx = fmaxf(mx, __shfl_xor(mx, 16, 32));

        float sum = 0.0f;
        for (int j = 0; j < S_LEN / 2; ++j) sum += __expf(bf16_to_f32(row[j]) - mx);
        sum += __shfl_xor(sum, 16, 32);
        const float inv = 1.0f / sum;

        float* arow = AttnMean + ((size_t)b * S_LEN + (q0 + r)) * S_LEN + c0;
        for (int j = 0; j < S_LEN / 2; ++j) {
            float p = __expf(bf16_to_f32(row[j]) - mx) * inv;
            row[j] = f32_to_bf16(p);                      // probs in place
            atomicAdd(arow + j, p * (1.0f / HEADS));      // head-mean accumulation
        }
    }
    __syncthreads();

    // ---- ctx = P @ V   (K-dim = S_LEN, N-dim = dk in 4 tiles of 16)
    f32x8 acc[4] = {};
    for (int ks = 0; ks < S_LEN / 32; ++ks) {
        const uint16_t* pa = sc + (size_t)mrow * S_LEN + ks * 32 + khalf * 8;
        bf16x16 af = make_frag(pa, pa + 16);              // LDS ds_load_b128 x2
#pragma unroll
        for (int dt = 0; dt < 4; ++dt) {
            const uint16_t* pb = Vt + ((size_t)(b * HEADS + h) * DKH + dt * 16 + mrow) * S_LEN
                                    + ks * 32 + khalf * 8;
            bf16x16 bfr = make_frag(pb, pb + 16);
            acc[dt] = wmma_bf16(af, bfr, acc[dt]);
        }
    }

#pragma unroll
    for (int dt = 0; dt < 4; ++dt)
#pragma unroll
        for (int v = 0; v < 8; ++v) {
            int row = q0 + khalf * 8 + v;
            int col = h * DKH + dt * 16 + mrow;
            Ctx[((size_t)row * BATCH + b) * DMODEL + col] = f32_to_bf16(acc[dt][v]);
        }
}

// ================================================================ host launcher
extern "C" void kernel_launch(void* const* d_in, const int* in_sizes, int n_in,
                              void* d_out, int out_size, void* d_ws, size_t ws_size,
                              hipStream_t stream) {
    const float* q_in = (const float*)d_in[0];
    const float* k_in = (const float*)d_in[1];
    const float* v_in = (const float*)d_in[2];
    const float* wq   = (const float*)d_in[3];
    const float* wk   = (const float*)d_in[4];
    const float* wv   = (const float*)d_in[5];
    const float* wo   = (const float*)d_in[6];

    float* out       = (float*)d_out;                                  // [S,B,D]
    float* attn_mean = out + (size_t)S_LEN * BATCH * DMODEL;           // [B,S,S]

    const size_t ACT = (size_t)S_LEN * BATCH * DMODEL;   // 4,194,304 elems
    const size_t WEL = (size_t)DMODEL * DMODEL;          // 1,048,576 elems

    uint16_t* p   = (uint16_t*)d_ws;
    uint16_t* Xq  = p;  p += ACT;
    uint16_t* Xk  = p;  p += ACT;
    uint16_t* Xv  = p;  p += ACT;
    uint16_t* Wqb = p;  p += WEL;
    uint16_t* Wkb = p;  p += WEL;
    uint16_t* Wvb = p;  p += WEL;
    uint16_t* Wob = p;  p += WEL;
    uint16_t* Qb  = p;  p += ACT;
    uint16_t* Kb  = p;  p += ACT;
    uint16_t* Vb  = p;  p += ACT;
    uint16_t* Vt  = p;  p += ACT;
    uint16_t* Ctx = p;  p += ACT;

    // 1) casts to bf16
    cast_f32_to_bf16<<<2048, 256, 0, stream>>>(q_in, Xq, (int)ACT);
    cast_f32_to_bf16<<<2048, 256, 0, stream>>>(k_in, Xk, (int)ACT);
    cast_f32_to_bf16<<<2048, 256, 0, stream>>>(v_in, Xv, (int)ACT);
    cast_f32_to_bf16<<<1024, 256, 0, stream>>>(wq, Wqb, (int)WEL);
    cast_f32_to_bf16<<<1024, 256, 0, stream>>>(wk, Wkb, (int)WEL);
    cast_f32_to_bf16<<<1024, 256, 0, stream>>>(wv, Wvb, (int)WEL);
    cast_f32_to_bf16<<<1024, 256, 0, stream>>>(wo, Wob, (int)WEL);

    // 2) QKV projections (bf16 WMMA GEMM, bf16 out); block tile 128x128, wave tile 64x64
    dim3 gg(ROWS / 128, DMODEL / 128);
    gemm_bf16_wmma<<<gg, 128, 0, stream>>>(Xq, Wqb, Qb, nullptr, ROWS, DMODEL, DMODEL, 0);
    gemm_bf16_wmma<<<gg, 128, 0, stream>>>(Xk, Wkb, Kb, nullptr, ROWS, DMODEL, DMODEL, 0);
    gemm_bf16_wmma<<<gg, 128, 0, stream>>>(Xv, Wvb, Vb, nullptr, ROWS, DMODEL, DMODEL, 0);

    // 3) transpose V for contiguous B-fragment loads in P@V
    transpose_v<<<2048, 256, 0, stream>>>(Vb, Vt);

    // 4) zero the head-averaged attention output (accumulated atomically)
    zero_f32<<<4096, 256, 0, stream>>>(attn_mean, BATCH * S_LEN * S_LEN);

    // 5) fused attention: 4 waves/block, 256KB LDS (bf16 score/prob strips)
    dim3 ga(S_LEN / 64, BATCH * HEADS);
    size_t lds = (size_t)4 * 16 * S_LEN * 2;   // 262144 B
    attention_wmma<<<ga, 128, lds, stream>>>(Qb, Kb, Vt, Ctx, attn_mean);

    // 6) output projection, f32 result straight into d_out
    gemm_bf16_wmma<<<gg, 128, 0, stream>>>(Ctx, Wob, nullptr, out, ROWS, DMODEL, DMODEL, 1);
}
